// MultiHeadAttention_54992761258328
// MI455X (gfx1250) — matrix operand order repaired and checked by reading, not compiled, not run on hardware
//
#include <hip/hip_runtime.h>

// MI455X / gfx1250 multi-head attention forward.
// bf16 WMMA (v_wmma_f32_16x16x32_bf16) everywhere, fp32 accumulate.
// K-tile staging in attention uses GLOBAL_LOAD_ASYNC_TO_LDS_B128 (ASYNCcnt).
// B=2, S=2048, D=1024, H=16, DK=64.

typedef __bf16 bf16;
typedef bf16  v16bf __attribute__((ext_vector_type(16)));
typedef bf16  v8bf  __attribute__((ext_vector_type(8)));
typedef float v8f   __attribute__((ext_vector_type(8)));

#define NB     2
#define SEQ    2048
#define DMODEL 1024
#define NH     16
#define DKH    64
#define MROWS  (NB * SEQ)   // 4096

// D = A*B + C, bf16 inputs, f32 accumulate, 16x16x32
__device__ inline v8f wmma_bf16(v16bf a, v16bf b, v8f c) {
  return __builtin_amdgcn_wmma_f32_16x16x32_bf16(
      /*neg_a=*/false, a, /*neg_b=*/false, b,
      /*c_mod=*/(short)0, c, /*reuse_a=*/false, /*reuse_b=*/false);
}

__device__ inline void cvt8(v16bf& f, int base, float4 a, float4 b) {
  f[base + 0] = (bf16)a.x; f[base + 1] = (bf16)a.y;
  f[base + 2] = (bf16)a.z; f[base + 3] = (bf16)a.w;
  f[base + 4] = (bf16)b.x; f[base + 5] = (bf16)b.y;
  f[base + 6] = (bf16)b.z; f[base + 7] = (bf16)b.w;
}

// A fragment (16x32 bf16, M x K) from an fp32 row.
// Layout used here: lane(l,hi) holds row M=l, K = {k0+16hi..+15} (contiguous, matches B).
__device__ inline v16bf load_a_frag_f32(const float* __restrict__ row, int k0, int hi) {
  const float4* p0 = (const float4*)(row + k0 + 16 * hi);
  const float4* p1 = (const float4*)(row + k0 + 16 * hi + 8);
  v16bf f;
  cvt8(f, 0, p0[0], p0[1]);
  cvt8(f, 8, p1[0], p1[1]);
  return f;
}

// Same A-fragment pattern from a bf16 row (two 8-element contiguous runs).
__device__ inline v16bf load_a_frag_bf16(const bf16* __restrict__ row, int k0, int hi) {
  v8bf r0 = *(const v8bf*)(row + k0 + 16 * hi);
  v8bf r1 = *(const v8bf*)(row + k0 + 16 * hi + 8);
  v16bf f;
#pragma unroll
  for (int i = 0; i < 8; i++) { f[i] = r0[i]; f[8 + i] = r1[i]; }
  return f;
}

// B fragment (32x16 bf16, K x N) from fp32: lane(l,hi) holds col N=l,
// K = 16hi..16hi+15 -> 16 consecutive elements at p.
__device__ inline v16bf load_b_frag_f32(const float* __restrict__ p) {
  const float4* q = (const float4*)p;
  v16bf f;
  cvt8(f, 0, q[0], q[1]);
  cvt8(f, 8, q[2], q[3]);
  return f;
}

// ---------------------------------------------------------------------------
// Kernel 1: y = x @ W^T + b  ->  bf16 head-major [B,H,S,DK]
// grid (M/16, 2, 3), block 256 (8 waves); wave strip = 16 rows x 64 cols.
// ---------------------------------------------------------------------------
__global__ __launch_bounds__(256, 2) void qkv_proj_kernel(
    const float* __restrict__ q, const float* __restrict__ k, const float* __restrict__ v,
    const float* __restrict__ Wq, const float* __restrict__ Wk, const float* __restrict__ Wv,
    const float* __restrict__ bq, const float* __restrict__ bk, const float* __restrict__ bv,
    bf16* __restrict__ qh, bf16* __restrict__ kh, bf16* __restrict__ vh) {
  const int z = blockIdx.z;
  const float* X    = (z == 0) ? q  : (z == 1) ? k  : v;
  const float* W    = (z == 0) ? Wq : (z == 1) ? Wk : Wv;
  const float* bias = (z == 0) ? bq : (z == 1) ? bk : bv;
  bf16* Y           = (z == 0) ? qh : (z == 1) ? kh : vh;

  const int tid  = threadIdx.x;
  const int lane = tid & 31;
  const int w    = tid >> 5;
  const int l    = lane & 15;
  const int hi   = lane >> 4;

  const int m0 = blockIdx.x * 16;
  const int n0 = (blockIdx.y * 8 + w) * 64;

  v8f acc[4] = {};
  const float* Xrow = X + (size_t)(m0 + l) * DMODEL;

  for (int k0 = 0; k0 < DMODEL; k0 += 32) {
    v16bf af = load_a_frag_f32(Xrow, k0, hi);
#pragma unroll
    for (int i = 0; i < 4; i++) {
      const float* wrow = W + (size_t)(n0 + i * 16 + l) * DMODEL + k0 + 16 * hi;
      acc[i] = wmma_bf16(af, load_b_frag_f32(wrow), acc[i]);
    }
  }

#pragma unroll
  for (int i = 0; i < 4; i++) {
    const int n  = n0 + i * 16 + l;
    const float bs = bias[n];
    const int h  = n >> 6;     // head
    const int dk = n & 63;
#pragma unroll
    for (int r = 0; r < 8; r++) {
      const int m = m0 + r + 8 * hi;      // C layout: row = r + 8*hi
      const int b = m >> 11;              // /SEQ
      const int s = m & (SEQ - 1);
      Y[(((size_t)b * NH + h) * SEQ + s) * DKH + dk] = (bf16)(acc[i][r] + bs);
    }
  }
}

// ---------------------------------------------------------------------------
// Kernel 2: flash attention per (b,h). grid (B*H, S/128), block 256.
// Wave w owns q-rows [q0, q0+16). 32-key blocks staged in LDS.
// K block staged with GLOBAL_LOAD_ASYNC_TO_LDS_B128 (no register pass).
// ---------------------------------------------------------------------------
__global__ __launch_bounds__(256, 2) void attn_kernel(
    const bf16* __restrict__ qh, const bf16* __restrict__ kh, const bf16* __restrict__ vh,
    const int* __restrict__ mask, bf16* __restrict__ ctx) {
  __shared__ bf16 Ksh[32 * 64];      // [key][dk]
  __shared__ bf16 VshT[64 * 32];     // [dk][key]  (transposed V block)
  __shared__ bf16 Psh[8][16 * 32];   // per-wave P transpose scratch

  const int tid  = threadIdx.x;
  const int lane = tid & 31;
  const int w    = tid >> 5;
  const int l    = lane & 15;
  const int hi   = lane >> 4;

  const int bh = blockIdx.x;               // b*H + h
  const int b  = bh >> 4;
  const int h  = bh & (NH - 1);
  const int q0 = blockIdx.y * 128 + w * 16;

  const size_t head_base = (size_t)bh * SEQ * DKH;
  const bf16* qhead = qh + head_base;
  const bf16* khead = kh + head_base;
  const bf16* vhead = vh + head_base;
  const int*  maskb = mask + (size_t)b * SEQ * SEQ;

  // LDS byte address of Ksh for this thread's async-copy slot.
  // Generic LDS address low 32 bits == wave-logical LDS byte address (ISA 10.2).
  const unsigned ks_lds = (unsigned)(uintptr_t)(&Ksh[0]) + (unsigned)(tid * 16);

  // Q fragments for this wave's 16 rows, whole DK=64: two K=32 fragments.
  const bf16* qrow = qhead + (size_t)(q0 + l) * DKH;
  const v16bf qf0 = load_a_frag_bf16(qrow, 0, hi);
  const v16bf qf1 = load_a_frag_bf16(qrow, 32, hi);

  v8f acc[4] = {};
  float mrun[8], rsum[8];
#pragma unroll
  for (int r = 0; r < 8; r++) { mrun[r] = -1e30f; rsum[r] = 0.0f; }

  for (int kb = 0; kb < SEQ; kb += 32) {
    // ---- K block [32][64] bf16 = 4KB: async copy straight into LDS ----
    // GVS mode: addr = SGPR64 (khead + kb rows) + VGPR32 (tid*16 bytes)
    {
      const void* kbase = (const void*)(khead + (size_t)kb * DKH);
      const unsigned voff = (unsigned)(tid * 16);
      asm volatile("global_load_async_to_lds_b128 %0, %1, %2"
                   :
                   : "v"(ks_lds), "v"(voff), "s"(kbase)
                   : "memory");
    }

    // ---- V block: register pass (needs transpose) -> VshT [64][32] ----
    {
      const int e0  = tid * 8;          // 2048 elems / 256 threads
      const int key = e0 >> 6;
      const int dk  = e0 & 63;
      v8bf vv = *(const v8bf*)(vhead + (size_t)(kb + key) * DKH + dk);
#pragma unroll
      for (int i = 0; i < 8; i++) VshT[(dk + i) * 32 + key] = vv[i];
    }

    // prefetch next block while we compute this one
    if (kb + 32 < SEQ) {
      __builtin_prefetch(khead + (size_t)(kb + 32) * DKH + tid * 8, 0, 0);
      __builtin_prefetch(vhead + (size_t)(kb + 32) * DKH + tid * 8, 0, 0);
    }

    // all async K-copies from this wave complete, then block-wide barrier
    asm volatile("s_wait_asynccnt 0x0" ::: "memory");
    __syncthreads();

    // ---- scores: two 16x16 tiles over keys [kb, kb+32) ----
    v8f sc[2] = {};
#pragma unroll
    for (int kt = 0; kt < 2; kt++) {
      const bf16* krow = &Ksh[(kt * 16 + l) * 64];
      v16bf bk0 = *(const v16bf*)(krow + 16 * hi);        // dk 0..31 window
      v16bf bk1 = *(const v16bf*)(krow + 32 + 16 * hi);   // dk 32..63 window
      sc[kt] = wmma_bf16(qf0, bk0, sc[kt]);
      sc[kt] = wmma_bf16(qf1, bk1, sc[kt]);
    }

    // ---- mask + scale + online softmax over these 32 columns ----
#pragma unroll
    for (int r = 0; r < 8; r++) {
      const int row = q0 + r + 8 * hi;
      const int* mrow = maskb + (size_t)row * SEQ + kb;
      float s0 = mrow[l]      ? sc[0][r] * 0.125f : -1e-9f;  // /sqrt(64); (sic) fill
      float s1 = mrow[16 + l] ? sc[1][r] * 0.125f : -1e-9f;

      float tm = fmaxf(s0, s1);
      tm = fmaxf(tm, __shfl_xor(tm, 1));
      tm = fmaxf(tm, __shfl_xor(tm, 2));
      tm = fmaxf(tm, __shfl_xor(tm, 4));
      tm = fmaxf(tm, __shfl_xor(tm, 8));
      const float mnew = fmaxf(mrun[r], tm);

      const float p0 = __expf(s0 - mnew);
      const float p1 = __expf(s1 - mnew);
      float ts = p0 + p1;
      ts += __shfl_xor(ts, 1);
      ts += __shfl_xor(ts, 2);
      ts += __shfl_xor(ts, 4);
      ts += __shfl_xor(ts, 8);

      const float sold = __expf(mrun[r] - mnew);
      rsum[r] = rsum[r] * sold + ts;
      mrun[r] = mnew;
#pragma unroll
      for (int i = 0; i < 4; i++) acc[i][r] *= sold;

      // stash P (bf16) for in-wave transpose: C layout -> A layout via LDS
      Psh[w][(r + 8 * hi) * 32 + l]      = (bf16)p0;
      Psh[w][(r + 8 * hi) * 32 + 16 + l] = (bf16)p1;
    }

    // ---- PV: P(16x32) x V(32x64) -> 4 accumulators ----
    {
      const bf16* prow = &Psh[w][l * 32];
      const v16bf pf = load_a_frag_bf16(prow, 0, hi);
#pragma unroll
      for (int i = 0; i < 4; i++) {
        const bf16* vrow = &VshT[(i * 16 + l) * 32 + 16 * hi];
        v16bf vf = *(const v16bf*)vrow;
        acc[i] = wmma_bf16(pf, vf, acc[i]);
      }
    }
    __syncthreads();
  }

  // ---- normalize and write ctx as bf16 [B,S,D] ----
#pragma unroll
  for (int i = 0; i < 4; i++) {
#pragma unroll
    for (int r = 0; r < 8; r++) {
      const int row = q0 + r + 8 * hi;
      const float o = acc[i][r] / rsum[r];
      ctx[((size_t)b * SEQ + row) * DMODEL + h * DKH + i * 16 + l] = (bf16)o;
    }
  }
}

// ---------------------------------------------------------------------------
// Kernel 3: out = ctx @ Wo^T + bo  (fp32 output). grid (M/16, 2), block 256.
// ---------------------------------------------------------------------------
__global__ __launch_bounds__(256, 2) void out_proj_kernel(
    const bf16* __restrict__ ctx, const float* __restrict__ Wo,
    const float* __restrict__ bo, float* __restrict__ out) {
  const int tid  = threadIdx.x;
  const int lane = tid & 31;
  const int w    = tid >> 5;
  const int l    = lane & 15;
  const int hi   = lane >> 4;

  const int m0 = blockIdx.x * 16;
  const int n0 = (blockIdx.y * 8 + w) * 64;

  v8f acc[4] = {};
  const bf16* arow = ctx + (size_t)(m0 + l) * DMODEL;

  for (int k0 = 0; k0 < DMODEL; k0 += 32) {
    v16bf af = load_a_frag_bf16(arow, k0, hi);
#pragma unroll
    for (int i = 0; i < 4; i++) {
      const float* wrow = Wo + (size_t)(n0 + i * 16 + l) * DMODEL + k0 + 16 * hi;
      acc[i] = wmma_bf16(af, load_b_frag_f32(wrow), acc[i]);
    }
  }

#pragma unroll
  for (int i = 0; i < 4; i++) {
    const int n = n0 + i * 16 + l;
    const float bs = bo[n];
#pragma unroll
    for (int r = 0; r < 8; r++) {
      const int m = m0 + r + 8 * hi;
      out[(size_t)m * DMODEL + n] = acc[i][r] + bs;
    }
  }
}

// ---------------------------------------------------------------------------
extern "C" void kernel_launch(void* const* d_in, const int* in_sizes, int n_in,
                              void* d_out, int out_size, void* d_ws, size_t ws_size,
                              hipStream_t stream) {
  const float* q    = (const float*)d_in[0];
  const float* k    = (const float*)d_in[1];
  const float* v    = (const float*)d_in[2];
  const int*   mask = (const int*)d_in[3];
  const float* Wq   = (const float*)d_in[4];
  const float* bq   = (const float*)d_in[5];
  const float* Wk   = (const float*)d_in[6];
  const float* bk   = (const float*)d_in[7];
  const float* Wv   = (const float*)d_in[8];
  const float* bv   = (const float*)d_in[9];
  const float* Wo   = (const float*)d_in[10];
  const float* bo   = (const float*)d_in[11];
  float* out = (float*)d_out;

  char* ws = (char*)d_ws;
  const size_t HEAD_BYTES = (size_t)NB * NH * SEQ * DKH * sizeof(bf16);  // 8 MB
  bf16* qh  = (bf16*)(ws + 0 * HEAD_BYTES);
  bf16* kh  = (bf16*)(ws + 1 * HEAD_BYTES);
  bf16* vh  = (bf16*)(ws + 2 * HEAD_BYTES);
  bf16* ctx = (bf16*)(ws + 3 * HEAD_BYTES);

  dim3 blk(256, 1, 1);
  qkv_proj_kernel<<<dim3(MROWS / 16, 2, 3), blk, 0, stream>>>(
      q, k, v, Wq, Wk, Wv, bq, bk, bv, qh, kh, vh);
  attn_kernel<<<dim3(NB * NH, SEQ / 128, 1), blk, 0, stream>>>(qh, kh, vh, mask, ctx);
  out_proj_kernel<<<dim3(MROWS / 16, 2, 1), blk, 0, stream>>>(ctx, Wo, bo, out);
}
